// Model_68539088110301
// MI455X (gfx1250) — matrix operand order repaired and checked by reading, not compiled, not run on hardware
//
#include <hip/hip_runtime.h>
#include <stdint.h>

// ---------------------------------------------------------------------------
// MI455X (gfx1250, wave32) implementation.
// GEMMs (~280 GFLOP total) run through v_wmma_f32_16x16x32_f16 with f32
// accumulation. 128x128 block tile, double-buffered LDS, A prefetched via
// VGPRs (f32->f16 pack convert), B tiles DMA'd with global_load_async_to_lds
// (ASYNCcnt) straight into LDS in WMMA fragment layout.
// ---------------------------------------------------------------------------

typedef __attribute__((ext_vector_type(4)))  _Float16 v4h;
typedef __attribute__((ext_vector_type(8)))  _Float16 v8h;
typedef __attribute__((ext_vector_type(16))) _Float16 v16h;
typedef __attribute__((ext_vector_type(8)))  float    v8f;

#define LEAKY_SLOPE 0.01f

static __device__ __forceinline__ void wait_async0() {
#if __has_builtin(__builtin_amdgcn_s_wait_asynccnt)
  __builtin_amdgcn_s_wait_asynccnt(0);
#else
  asm volatile("s_wait_asynccnt 0x0" ::: "memory");
#endif
}

// ---------------------------------------------------------------------------
// WMMA GEMM:  C[M,N] = act( A[M,Kpad](f32) * B[Kpad,N] + bias[N] )
// B pre-transposed + converted: Bt[N][Kpad] f16 (zero padded).
// Block tile 128x128, K-step 32, 256 threads = 8 waves (4 down M x 2 across
// N); each wave computes a 32x64 patch as 2x4 v_wmma_f32_16x16x32_f16 tiles.
// Double-buffered LDS; B tiles via async-to-LDS DMA.
// Requires: M % 128 == 0, N % 128 == 0, Kpad % 32 == 0.
// ACT: 0 = relu, 1 = leaky relu.
// ---------------------------------------------------------------------------
template <int ACT>
__global__ __launch_bounds__(256) void wmma_gemm_bias_act(
    const float* __restrict__ A, int ldA,
    const _Float16* __restrict__ Bt, int Kpad,
    const float* __restrict__ bias,
    float* __restrict__ C, int ldC) {
  __shared__ _Float16 As[2][128][40];   // 32 cols + 8 pad (16B-aligned rows)
  __shared__ _Float16 Bs[2][128][40];

  const int tid  = threadIdx.x;
  const int lane = tid & 31;
  const int wave = tid >> 5;
  const int wr   = wave & 3;     // wave row (M): 4 x 32
  const int wc   = wave >> 2;    // wave col (N): 2 x 64
  const int lmod = lane & 15;
  const int lhi  = lane >> 4;
  const int m0 = blockIdx.x * 128;
  const int n0 = blockIdx.y * 128;

  const unsigned long long bbase = (unsigned long long)(uintptr_t)Bt;

  float4 pa[4];  // A tile prefetch registers (128x32 f32 / 256 threads)

  // load A tile (k0) into registers
  auto loadA = [&](int k0) {
#pragma unroll
    for (int i = 0; i < 4; ++i) {
      int s   = tid + i * 256;
      int row = s >> 3;
      int c4  = (s & 7) << 2;
      pa[i] = *(const float4*)(A + (size_t)(m0 + row) * ldA + k0 + c4);
    }
  };
  // convert + store prefetched A registers into LDS buffer
  auto storeA = [&](int buf) {
#pragma unroll
    for (int i = 0; i < 4; ++i) {
      int s   = tid + i * 256;
      int row = s >> 3;
      int c4  = (s & 7) << 2;
      v4h h;
      h[0] = (_Float16)pa[i].x; h[1] = (_Float16)pa[i].y;
      h[2] = (_Float16)pa[i].z; h[3] = (_Float16)pa[i].w;
      *(v4h*)&As[buf][row][c4] = h;
    }
  };
  // B tile (128 x 32 f16 = 8KB): async DMA global->LDS, 16B per lane per op
  auto issueB = [&](int k0, int buf) {
#pragma unroll
    for (int i = 0; i < 2; ++i) {
      int c   = tid + i * 256;      // 512 16B chunks
      int row = c >> 2;
      int ch  = (c & 3) << 3;       // halves
      uint32_t lds = (uint32_t)(uintptr_t)&Bs[buf][row][ch];
      uint32_t go  = (uint32_t)(((size_t)(n0 + row) * Kpad + k0 + ch) * 2);
      asm volatile("global_load_async_to_lds_b128 %0, %1, %2"
                   :: "v"(lds), "v"(go), "s"(bbase)
                   : "memory");
    }
  };

  v8f acc[2][4] = {};

  // prologue: stage k0 = 0 into buffer 0
  loadA(0);
  issueB(0, 0);
  storeA(0);
  wait_async0();
  __syncthreads();

  int buf = 0;
  for (int k0 = 0; k0 < Kpad; k0 += 32) {
    const int knext = k0 + 32;
    const bool more = knext < Kpad;
    if (more) {            // overlap next tile's fetch with this tile's math
      loadA(knext);
      issueB(knext, buf ^ 1);
    }

    // fragments per ISA 7.12.2 (wave32 16-bit layouts)
    v16h afrag[2], bfrag[4];
#pragma unroll
    for (int mi = 0; mi < 2; ++mi) {
      int r = wr * 32 + mi * 16 + lmod;
      v8h lo = *(const v8h*)&As[buf][r][lhi * 8];       // K = hi*8 .. +7
      v8h hi = *(const v8h*)&As[buf][r][16 + lhi * 8];  // K = 16+hi*8 .. +7
      afrag[mi] = __builtin_shufflevector(lo, hi,
          0,1,2,3,4,5,6,7,8,9,10,11,12,13,14,15);
    }
#pragma unroll
    for (int ni = 0; ni < 4; ++ni) {
      int r = wc * 64 + ni * 16 + lmod;                 // N column
      v8h lo = *(const v8h*)&Bs[buf][r][lhi * 8];       // K = hi*8 .. +7
      v8h hi = *(const v8h*)&Bs[buf][r][16 + lhi * 8];  // K = 16+hi*8 .. +7
      bfrag[ni] = __builtin_shufflevector(lo, hi,
          0,1,2,3,4,5,6,7,8,9,10,11,12,13,14,15);
    }
#pragma unroll
    for (int mi = 0; mi < 2; ++mi)
#pragma unroll
      for (int ni = 0; ni < 4; ++ni)
        acc[mi][ni] = __builtin_amdgcn_wmma_f32_16x16x32_f16(
            false, afrag[mi], false, bfrag[ni], (short)0, acc[mi][ni],
            false, false);

    if (more) {
      storeA(buf ^ 1);     // A regs -> other LDS buffer
      wait_async0();       // next B tile resident
      __syncthreads();
      buf ^= 1;
    }
  }

  // epilogue: bias + activation (compile-time), C layout per 7.12.2
#pragma unroll
  for (int mi = 0; mi < 2; ++mi)
#pragma unroll
    for (int ni = 0; ni < 4; ++ni) {
      int n = n0 + wc * 64 + ni * 16 + lmod;
      float bv = bias[n];
#pragma unroll
      for (int r = 0; r < 8; ++r) {
        int m = m0 + wr * 32 + mi * 16 + lhi * 8 + r;
        float v = acc[mi][ni][r] + bv;
        if (ACT == 0) v = fmaxf(v, 0.0f);
        else          v = v >= 0.0f ? v : LEAKY_SLOPE * v;
        C[(size_t)m * ldC + n] = v;
      }
    }
}

// ---------------------------------------------------------------------------
// Weight convert: W[K,N] f32 -> Wt[N,Kpad] f16, zero-padded K..Kpad.
// ---------------------------------------------------------------------------
__global__ void wconv_kernel(const float* __restrict__ W,
                             _Float16* __restrict__ Wt,
                             int K, int N, int Kpad) {
  int idx = blockIdx.x * 256 + threadIdx.x;
  if (idx >= N * Kpad) return;
  int n = idx / Kpad, k = idx % Kpad;
  Wt[idx] = (k < K) ? (_Float16)W[(size_t)k * N + n] : (_Float16)0.0f;
}

// obj_vecs0 = concat(obj_emb[objs], attributes)  -> [8192, 163]
__global__ void obj0_kernel(const int* __restrict__ objs,
                            const float* __restrict__ obj_emb,
                            const float* __restrict__ attrs,
                            float* __restrict__ out) {
  int idx = blockIdx.x * 256 + threadIdx.x;
  if (idx >= 8192 * 163) return;
  int i = idx / 163, c = idx % 163;
  out[idx] = (c < 128) ? obj_emb[objs[i] * 128 + c] : attrs[i * 35 + (c - 128)];
}

// pred_vecs0 = pred_emb[p_idx] -> [32768, 128]
__global__ void pred_gather_kernel(const int* __restrict__ triples,
                                   const float* __restrict__ pred_emb,
                                   float* __restrict__ out, int T) {
  int idx = blockIdx.x * 256 + threadIdx.x;
  if (idx >= T * 128) return;
  int t = idx >> 7, c = idx & 127;
  out[idx] = pred_emb[triples[t * 3 + 1] * 128 + c];
}

// t = [obj[s] | pred | obj[o] | zero-pad] with ld columns
__global__ void build_t_kernel(float* __restrict__ t, int ld,
                               const float* __restrict__ objv, int obj_dim,
                               const float* __restrict__ pred, int pred_ld,
                               int pred_off,
                               const int* __restrict__ triples, int T) {
  long long idx = (long long)blockIdx.x * 256 + threadIdx.x;
  if (idx >= (long long)T * ld) return;
  int row = (int)(idx / ld), col = (int)(idx % ld);
  float v;
  if (col < obj_dim) {
    v = objv[(size_t)triples[row * 3 + 0] * obj_dim + col];
  } else if (col < obj_dim + 128) {
    v = pred[(size_t)row * pred_ld + pred_off + (col - obj_dim)];
  } else if (col < 2 * obj_dim + 128) {
    v = objv[(size_t)triples[row * 3 + 2] * obj_dim + (col - obj_dim - 128)];
  } else {
    v = 0.0f;
  }
  t[idx] = v;
}

__global__ void zero_kernel(float* __restrict__ p, int n) {
  int idx = blockIdx.x * 256 + threadIdx.x;
  if (idx < n) p[idx] = 0.0f;
}

// pooled[s] += new_t[:, :512]; pooled[o] += new_t[:, 640:1152]
__global__ void scatter_kernel(const float* __restrict__ new_t,
                               const int* __restrict__ triples,
                               float* __restrict__ pooled, int T) {
  long long idx = (long long)blockIdx.x * 256 + threadIdx.x;
  if (idx >= (long long)T * 512) return;
  int row = (int)(idx >> 9), c = (int)(idx & 511);
  int s = triples[row * 3 + 0], o = triples[row * 3 + 2];
  atomicAdd(&pooled[(size_t)s * 512 + c], new_t[(size_t)row * 1152 + c]);
  atomicAdd(&pooled[(size_t)o * 512 + c], new_t[(size_t)row * 1152 + 640 + c]);
}

__global__ void counts_kernel(const int* __restrict__ triples,
                              float* __restrict__ counts, int T) {
  int t = blockIdx.x * 256 + threadIdx.x;
  if (t >= T) return;
  atomicAdd(&counts[triples[t * 3 + 0]], 1.0f);
  atomicAdd(&counts[triples[t * 3 + 2]], 1.0f);
}

__global__ void divide_kernel(float* __restrict__ pooled,
                              const float* __restrict__ counts) {
  int idx = blockIdx.x * 256 + threadIdx.x;
  if (idx >= 8192 * 512) return;
  pooled[idx] /= fmaxf(counts[idx >> 9], 1.0f);
}

// direct stride-2 VALID 3x3 conv + fused batchnorm + leaky relu
__global__ void conv_bn_leaky_kernel(
    const float* __restrict__ x, const float* __restrict__ w,
    const float* __restrict__ b, const float* __restrict__ gamma,
    const float* __restrict__ beta, const float* __restrict__ mean,
    const float* __restrict__ var, float* __restrict__ y,
    int N, int Cin, int Hin, int Win, int Cout, int Hout, int Wout) {
  long long idx = (long long)blockIdx.x * 256 + threadIdx.x;
  long long total = (long long)N * Cout * Hout * Wout;
  if (idx >= total) return;
  int ox = (int)(idx % Wout);
  long long tmp = idx / Wout;
  int oy = (int)(tmp % Hout); tmp /= Hout;
  int co = (int)(tmp % Cout);
  int n  = (int)(tmp / Cout);
  float acc = 0.0f;
  const float* xb = x + (size_t)n * Cin * Hin * Win;
  const float* wb = w + (size_t)co * Cin * 9;
  for (int ci = 0; ci < Cin; ++ci) {
    const float* xc = xb + (size_t)ci * Hin * Win + (2 * oy) * Win + 2 * ox;
    const float* wc = wb + ci * 9;
#pragma unroll
    for (int ky = 0; ky < 3; ++ky)
#pragma unroll
      for (int kx = 0; kx < 3; ++kx)
        acc = fmaf(xc[ky * Win + kx], wc[ky * 3 + kx], acc);
  }
  float v = acc + b[co];
  v = (v - mean[co]) * rsqrtf(var[co] + 1e-5f);
  v = v * gamma[co] + beta[co];
  y[idx] = v >= 0.0f ? v : LEAKY_SLOPE * v;
}

__global__ void meanpool_kernel(const float* __restrict__ x,
                                float* __restrict__ out, int NC, int HW) {
  int idx = blockIdx.x * 256 + threadIdx.x;
  if (idx >= NC) return;
  const float* p = x + (size_t)idx * HW;
  float s = 0.0f;
  for (int i = 0; i < HW; ++i) s += p[i];
  out[idx] = s / (float)HW;
}

// box input = concat(obj_vecs[8192x128], inside_vecs[obj_to_img][8192x256])
__global__ void boxcat_kernel(const float* __restrict__ objv,
                              const float* __restrict__ inside,
                              const int* __restrict__ o2i,
                              float* __restrict__ out) {
  int idx = blockIdx.x * 256 + threadIdx.x;
  if (idx >= 8192 * 384) return;
  int i = idx / 384, c = idx % 384;
  out[idx] = (c < 128) ? objv[(size_t)i * 128 + c]
                       : inside[(size_t)o2i[i] * 256 + (c - 128)];
}

// final tiny head: [8192,512] x [512,4] + b, leaky
__global__ void boxhead2_kernel(const float* __restrict__ H,
                                const float* __restrict__ W,
                                const float* __restrict__ b,
                                float* __restrict__ out) {
  int idx = blockIdx.x * 256 + threadIdx.x;
  if (idx >= 8192 * 4) return;
  int i = idx >> 2, j = idx & 3;
  float acc = b[j];
  const float* h = H + (size_t)i * 512;
  for (int k = 0; k < 512; ++k) acc = fmaf(h[k], W[k * 4 + j], acc);
  out[idx] = acc >= 0.0f ? acc : LEAKY_SLOPE * acc;
}

// ---------------------------------------------------------------------------
// Host orchestration
// ---------------------------------------------------------------------------
extern "C" void kernel_launch(void* const* d_in, const int* in_sizes, int n_in,
                              void* d_out, int out_size, void* d_ws,
                              size_t ws_size, hipStream_t stream) {
  (void)in_sizes; (void)n_in; (void)out_size; (void)ws_size;
  const int O_N = 8192, T_N = 32768;

  // Input mapping (top-level insertion order; nested dicts in sorted-key
  // jax-flatten order):
  //  0 objs, 1 triples, 2 obj_to_img, 3 attributes, 4 boundary,
  //  5 obj_emb, 6 pred_emb,
  //  7 + L*8 + {0:W1,1:b1,2:W2,3:b2,4:W3,5:b3,6:W4,7:b4}  (gconv, 5 layers)
  //  47 + i*6 + {0:b,1:beta,2:gamma,3:mean,4:var,5:w}     (cnn, 4 layers)
  //  71:boxW1, 72:boxb1, 73:boxW2, 74:boxb2
  const int*   objs       = (const int*)d_in[0];
  const int*   triples    = (const int*)d_in[1];
  const int*   obj_to_img = (const int*)d_in[2];
  const float* attributes = (const float*)d_in[3];
  const float* boundary   = (const float*)d_in[4];
  const float* obj_emb    = (const float*)d_in[5];
  const float* pred_emb   = (const float*)d_in[6];

  auto blocks = [](long long total) -> unsigned {
    return (unsigned)((total + 255) / 256);
  };

  size_t off = 0;
  auto alloc = [&](size_t bytes) -> char* {
    off = (off + 255) & ~(size_t)255;
    char* p = (char*)d_ws + off;
    off += bytes;
    return p;
  };

  // f16 transposed weights
  _Float16 *wt1[5], *wt2[5], *wt3[5], *wt4[5];
  for (int L = 0; L < 5; ++L) {
    int kp = (L == 0) ? 480 : 384;
    wt1[L] = (_Float16*)alloc((size_t)512 * kp * 2);
    wt2[L] = (_Float16*)alloc((size_t)1152 * 512 * 2);
    wt3[L] = (_Float16*)alloc((size_t)512 * 512 * 2);
    wt4[L] = (_Float16*)alloc((size_t)128 * 512 * 2);
  }
  _Float16* wtbox = (_Float16*)alloc((size_t)512 * 384 * 2);

  // activations (f32)
  float* obj0   = (float*)alloc((size_t)O_N * 163 * 4);
  float* objA   = (float*)alloc((size_t)O_N * 128 * 4);
  float* objB   = (float*)alloc((size_t)O_N * 128 * 4);
  float* pred0  = (float*)alloc((size_t)T_N * 128 * 4);
  float* tbuf   = (float*)alloc((size_t)T_N * 480 * 4);
  float* h1     = (float*)alloc((size_t)T_N * 512 * 4);
  float* newt   = (float*)alloc((size_t)T_N * 1152 * 4);
  float* pooled = (float*)alloc((size_t)O_N * 512 * 4);
  float* counts = (float*)alloc((size_t)O_N * 4);
  float* h2     = (float*)alloc((size_t)O_N * 512 * 4);
  float* y1     = (float*)alloc((size_t)64 * 32 * 63 * 63 * 4);
  float* y2     = (float*)alloc((size_t)64 * 64 * 31 * 31 * 4);
  float* y3     = (float*)alloc((size_t)64 * 128 * 15 * 15 * 4);
  float* y4     = (float*)alloc((size_t)64 * 256 * 7 * 7 * 4);
  float* inside = (float*)alloc((size_t)64 * 256 * 4);
  float* boxin  = (float*)alloc((size_t)O_N * 384 * 4);
  float* boxh   = (float*)alloc((size_t)O_N * 512 * 4);

  // ---- convert all GEMM weights to f16 [N][Kpad] ----
  for (int L = 0; L < 5; ++L) {
    int base = 7 + L * 8;
    int K1 = (L == 0) ? 454 : 384;
    int kp = (L == 0) ? 480 : 384;
    wconv_kernel<<<blocks((long long)512 * kp), 256, 0, stream>>>(
        (const float*)d_in[base + 0], wt1[L], K1, 512, kp);
    wconv_kernel<<<blocks((long long)1152 * 512), 256, 0, stream>>>(
        (const float*)d_in[base + 2], wt2[L], 512, 1152, 512);
    wconv_kernel<<<blocks((long long)512 * 512), 256, 0, stream>>>(
        (const float*)d_in[base + 4], wt3[L], 512, 512, 512);
    wconv_kernel<<<blocks((long long)128 * 512), 256, 0, stream>>>(
        (const float*)d_in[base + 6], wt4[L], 512, 128, 512);
  }
  wconv_kernel<<<blocks((long long)512 * 384), 256, 0, stream>>>(
      (const float*)d_in[71], wtbox, 384, 512, 384);

  // ---- initial gathers ----
  obj0_kernel<<<blocks((long long)O_N * 163), 256, 0, stream>>>(
      objs, obj_emb, attributes, obj0);
  pred_gather_kernel<<<blocks((long long)T_N * 128), 256, 0, stream>>>(
      triples, pred_emb, pred0, T_N);

  // ---- 5 graph-conv layers ----
  const float* objv = obj0;  int obj_dim = 163;
  const float* pred_src = pred0;  int pred_ld = 128, pred_off = 0;
  float* obj_out = objA;

  for (int L = 0; L < 5; ++L) {
    int base = 7 + L * 8;
    int kp = (L == 0) ? 480 : 384;

    build_t_kernel<<<blocks((long long)T_N * kp), 256, 0, stream>>>(
        tbuf, kp, objv, obj_dim, pred_src, pred_ld, pred_off, triples, T_N);

    wmma_gemm_bias_act<0><<<dim3(T_N / 128, 512 / 128), 256, 0, stream>>>(
        tbuf, kp, wt1[L], kp, (const float*)d_in[base + 1], h1, 512);
    wmma_gemm_bias_act<0><<<dim3(T_N / 128, 1152 / 128), 256, 0, stream>>>(
        h1, 512, wt2[L], 512, (const float*)d_in[base + 3], newt, 1152);

    zero_kernel<<<blocks((long long)O_N * 512), 256, 0, stream>>>(
        pooled, O_N * 512);
    zero_kernel<<<blocks((long long)O_N), 256, 0, stream>>>(counts, O_N);
    scatter_kernel<<<blocks((long long)T_N * 512), 256, 0, stream>>>(
        newt, triples, pooled, T_N);
    counts_kernel<<<blocks((long long)T_N), 256, 0, stream>>>(
        triples, counts, T_N);
    divide_kernel<<<blocks((long long)O_N * 512), 256, 0, stream>>>(
        pooled, counts);

    wmma_gemm_bias_act<0><<<dim3(O_N / 128, 512 / 128), 256, 0, stream>>>(
        pooled, 512, wt3[L], 512, (const float*)d_in[base + 5], h2, 512);
    wmma_gemm_bias_act<0><<<dim3(O_N / 128, 128 / 128), 256, 0, stream>>>(
        h2, 512, wt4[L], 512, (const float*)d_in[base + 7], obj_out, 128);

    objv = obj_out; obj_dim = 128;
    obj_out = (obj_out == objA) ? objB : objA;
    pred_src = newt; pred_ld = 1152; pred_off = 512;   // next layer's pred
  }
  const float* obj_final = objv;

  // ---- CNN (4 stride-2 conv + BN + leaky) ----
  struct CD { int cin, hin, win, cout, hout, wout; };
  CD cds[4] = {{3, 128, 128, 32, 63, 63},
               {32, 63, 63, 64, 31, 31},
               {64, 31, 31, 128, 15, 15},
               {128, 15, 15, 256, 7, 7}};
  const float* cx = boundary;
  float* couts[4] = {y1, y2, y3, y4};
  for (int i = 0; i < 4; ++i) {
    int cb = 47 + i * 6;  // b, beta, gamma, mean, var, w
    long long tot = (long long)64 * cds[i].cout * cds[i].hout * cds[i].wout;
    conv_bn_leaky_kernel<<<blocks(tot), 256, 0, stream>>>(
        cx, (const float*)d_in[cb + 5], (const float*)d_in[cb + 0],
        (const float*)d_in[cb + 2], (const float*)d_in[cb + 1],
        (const float*)d_in[cb + 3], (const float*)d_in[cb + 4],
        couts[i], 64, cds[i].cin, cds[i].hin, cds[i].win,
        cds[i].cout, cds[i].hout, cds[i].wout);
    cx = couts[i];
  }
  meanpool_kernel<<<blocks((long long)64 * 256), 256, 0, stream>>>(
      y4, inside, 64 * 256, 49);

  // ---- box head ----
  boxcat_kernel<<<blocks((long long)O_N * 384), 256, 0, stream>>>(
      obj_final, inside, obj_to_img, boxin);
  wmma_gemm_bias_act<1><<<dim3(O_N / 128, 512 / 128), 256, 0, stream>>>(
      boxin, 384, wtbox, 384, (const float*)d_in[72], boxh, 512);
  boxhead2_kernel<<<blocks((long long)O_N * 4), 256, 0, stream>>>(
      boxh, (const float*)d_in[73], (const float*)d_in[74], (float*)d_out);
}
